// AdaptiveConv1d_25142738550865
// MI455X (gfx1250) — matrix operand order repaired and checked by reading, not compiled, not run on hardware
//
#include <hip/hip_runtime.h>
#include <hip/hip_bf16.h>
#include <stdint.h>

typedef __bf16 bf16_t;
typedef __attribute__((ext_vector_type(8)))  bf16_t v8bf;
typedef __attribute__((ext_vector_type(16))) bf16_t v16bf;
typedef __attribute__((ext_vector_type(8)))  float  v8f;

#define B_   8
#define CIN  512
#define COUT 512
#define T_   8192
#define KTOT 1536   /* 3*CIN: [P | identity | F] stacked along K */
#define KSTEPS (KTOT / 32)

// ---------------------------------------------------------------------------
// CDNA5 async global->LDS copy (tracked by ASYNCcnt, no VGPR staging).
// Generic LDS pointer: low 32 bits == LDS byte offset (ISA 10.2 aperture rule).
// ---------------------------------------------------------------------------
__device__ __forceinline__ void async_copy_b128(void* lds_dst, const void* gsrc) {
  uint32_t lds_off = (uint32_t)(uintptr_t)lds_dst;
  asm volatile("global_load_async_to_lds_b128 %0, %1, off"
               :: "v"(lds_off), "v"((uint64_t)(uintptr_t)gsrc)
               : "memory");
}
__device__ __forceinline__ void wait_async0() {
  asm volatile("s_wait_asynccnt 0" ::: "memory");
}

// ---------------------------------------------------------------------------
// Kernel 1: per-(b,t) gather indices from dilation factors d (DILATION = 1).
// ---------------------------------------------------------------------------
__global__ void build_indices(const float* __restrict__ d,
                              int* __restrict__ idxP, int* __restrict__ idxF) {
  int i = blockIdx.x * blockDim.x + threadIdx.x;   // [0, B*T)
  if (i >= B_ * T_) return;
  int t = i & (T_ - 1);
  int dil = (int)d[i];          // trunc == floor (d > 0)
  if (dil < 1) dil = 1;
  int p = t - dil; if (p < 0) p = -p; p %= T_;
  int fraw = t + dil;
  int f = (fraw >= T_) ? (T_ - 1 - (fraw % T_)) : fraw;
  idxP[i] = p;
  idxF[i] = f;
}

// ---------------------------------------------------------------------------
// Kernel 2: transpose + convert  x[B,C,T] f32  ->  xT[B,T,C] bf16.
// ---------------------------------------------------------------------------
__global__ void transpose_convert(const float* __restrict__ x,
                                  bf16_t* __restrict__ xT) {
  __shared__ float tile[32][33];
  const int b  = blockIdx.z;
  const int c0 = blockIdx.y * 32;
  const int t0 = blockIdx.x * 32;
  const int tx = threadIdx.x & 31;
  const int ty = threadIdx.x >> 5;   // 0..7
  const float* xb = x + (size_t)b * CIN * T_;
#pragma unroll
  for (int j = 0; j < 4; ++j) {
    int c = c0 + ty * 4 + j;
    tile[ty * 4 + j][tx] = xb[(size_t)c * T_ + t0 + tx];
  }
  __syncthreads();
  bf16_t* xTb = xT + (size_t)b * T_ * CIN;
#pragma unroll
  for (int j = 0; j < 4; ++j) {
    int t = t0 + ty * 4 + j;
    xTb[(size_t)t * CIN + c0 + tx] = (bf16_t)tile[tx][ty * 4 + j];
  }
}

// ---------------------------------------------------------------------------
// Kernel 3: Wcat[COUT][3*CIN] bf16 = [W0 | W1 | W2]
// ---------------------------------------------------------------------------
__global__ void build_wcat(const float* __restrict__ W0,
                           const float* __restrict__ W1,
                           const float* __restrict__ W2,
                           bf16_t* __restrict__ Wcat) {
  int i = blockIdx.x * blockDim.x + threadIdx.x;
  if (i >= COUT * KTOT) return;
  int o = i / KTOT, k = i - o * KTOT;
  int seg = k >> 9, c = k & 511;
  const float* W = (seg == 0) ? W0 : (seg == 1) ? W1 : W2;
  Wcat[i] = (bf16_t)W[o * CIN + c];
}

// ---------------------------------------------------------------------------
// Kernel 4: fused gather-GEMM, double-buffered async-LDS pipeline.
// Block tile 128M x 128N, K-step 32; 8 waves as 4(M-groups of 32) x 2(N-groups
// of 64). Wave: 2 A-frags x 4 B-frags = 8 x v_wmma_f32_16x16x32_bf16 / K-step.
// LDS rows padded to 80 B (16B-aligned ds_load_b128, bank-spread).
// ---------------------------------------------------------------------------
__global__ __launch_bounds__(256)
void adaptive_conv_wmma(const bf16_t* __restrict__ xT,
                        const bf16_t* __restrict__ Wcat,
                        const float* __restrict__ bias,
                        const int* __restrict__ idxP,
                        const int* __restrict__ idxF,
                        float* __restrict__ out) {
  __shared__ __align__(16) bf16_t Asm[2][128 * 40];  // 2 x 10 KB
  __shared__ __align__(16) bf16_t Bsm[2][128 * 40];  // 2 x 10 KB

  const int b    = blockIdx.z;
  const int m0   = blockIdx.y * 128;
  const int n0   = blockIdx.x * 128;
  const int tid  = threadIdx.x;
  const int lane = tid & 31;
  const int wave = tid >> 5;
  const int wm   = wave & 3;          // M group (32 rows)
  const int wn   = wave >> 2;         // N group (64 cols)

  const bf16_t* xTb = xT + (size_t)b * T_ * CIN;
  const int* __restrict__ gP = idxP + b * T_;
  const int* __restrict__ gF = idxF + b * T_;

  // ---- loader precompute: 256 threads move 512 A-chunks + 512 B-chunks (16B)
  const int arow  = tid >> 2;          // 0..63 (and +64 for second chunk)
  const int apart = tid & 3;
  const bf16_t* aSrc0 = Wcat + (size_t)(m0 + arow) * KTOT + apart * 8;
  const bf16_t* aSrc1 = aSrc0 + (size_t)64 * KTOT;
  const int aOff = arow * 40 + apart * 8;

  const int bn    = tid >> 2;          // column 0..63 (and +64)
  const int bpart = tid & 3;
  const int t0g = n0 + bn, t1g = n0 + bn + 64;
  const bf16_t* bSrc0[3] = { xTb + (size_t)gP[t0g] * CIN + bpart * 8,
                             xTb + (size_t)t0g     * CIN + bpart * 8,
                             xTb + (size_t)gF[t0g] * CIN + bpart * 8 };
  const bf16_t* bSrc1[3] = { xTb + (size_t)gP[t1g] * CIN + bpart * 8,
                             xTb + (size_t)t1g     * CIN + bpart * 8,
                             xTb + (size_t)gF[t1g] * CIN + bpart * 8 };
  const int bOff = bn * 40 + bpart * 8;

  auto load_tiles = [&](int ks, int buf) {
    const int k0  = ks * 32;
    const int seg = k0 >> 9;           // 0:P  1:identity  2:F
    const int c0  = k0 & 511;
    async_copy_b128(&Asm[buf][aOff],            aSrc0 + k0);
    async_copy_b128(&Asm[buf][aOff + 64 * 40],  aSrc1 + k0);
    async_copy_b128(&Bsm[buf][bOff],            bSrc0[seg] + c0);
    async_copy_b128(&Bsm[buf][bOff + 64 * 40],  bSrc1[seg] + c0);
  };

  v8f acc[8] = {};                     // [a][f] -> acc[a*4+f]

  load_tiles(0, 0);
  wait_async0();
  __syncthreads();

  for (int ks = 0; ks < KSTEPS; ++ks) {
    const int buf = ks & 1;
    if (ks + 1 < KSTEPS) load_tiles(ks + 1, buf ^ 1);

    // A fragments (16-bit A 16x32 ISA layout):
    // lanes 0-15: row=lane, K0-7|K16-23 ; lanes 16-31: row=lane-16, K8-15|K24-31
    union { v16bf v; v8bf h[2]; } afr[2];
#pragma unroll
    for (int a = 0; a < 2; ++a) {
      int row = wm * 32 + a * 16 + (lane & 15);
      int off = (lane >> 4) * 8;
      afr[a].h[0] = *(const v8bf*)(&Asm[buf][row * 40 + off]);
      afr[a].h[1] = *(const v8bf*)(&Asm[buf][row * 40 + off + 16]);
    }
#pragma unroll
    for (int f = 0; f < 4; ++f) {
      // B fragment (32x16): lanes 0-15: col=lane, K0-7|K16-23; lanes 16-31: K8-15|K24-31
      union { v16bf v; v8bf h[2]; } bfr;
      int col = wn * 64 + f * 16 + (lane & 15);
      int off = (lane >> 4) * 8;
      bfr.h[0] = *(const v8bf*)(&Bsm[buf][col * 40 + off]);
      bfr.h[1] = *(const v8bf*)(&Bsm[buf][col * 40 + off + 16]);
#pragma unroll
      for (int a = 0; a < 2; ++a)
        acc[a * 4 + f] = __builtin_amdgcn_wmma_f32_16x16x32_bf16(
            false, afr[a].v, false, bfr.v, (short)0, acc[a * 4 + f], false, false);
    }

    if (ks + 1 < KSTEPS) wait_async0();  // new tile resident before barrier
    __syncthreads();
  }

  // ---- epilogue: bias + store.
  // C/D layout: VGPR r, lanes 0-15 -> M=r, lanes 16-31 -> M=r+8; N=lane&15.
  float* outb = out + (size_t)b * COUT * T_;
  const int mbase = m0 + wm * 32 + 8 * (lane >> 4);
  const int nbase = n0 + wn * 64 + (lane & 15);
#pragma unroll
  for (int a = 0; a < 2; ++a) {
#pragma unroll
    for (int f = 0; f < 4; ++f) {
      int t = nbase + f * 16;
#pragma unroll
      for (int r = 0; r < 8; ++r) {
        int o = mbase + a * 16 + r;
        outb[(size_t)o * T_ + t] = acc[a * 4 + f][r] + bias[o];
      }
    }
  }
}

// ---------------------------------------------------------------------------
extern "C" void kernel_launch(void* const* d_in, const int* in_sizes, int n_in,
                              void* d_out, int out_size, void* d_ws, size_t ws_size,
                              hipStream_t stream) {
  const float* x  = (const float*)d_in[0];   // [B, Cin, T]
  const float* d  = (const float*)d_in[1];   // [B, 1, T]
  const float* W0 = (const float*)d_in[2];   // [Cout, Cin]
  const float* b0 = (const float*)d_in[3];   // [Cout]
  const float* W1 = (const float*)d_in[4];
  const float* W2 = (const float*)d_in[5];
  float* out = (float*)d_out;                // [B, Cout, T]

  // workspace carve:
  //   [0, 256K)      idxP   int32 [B*T]
  //   [256K, 512K)   idxF   int32 [B*T]
  //   [512K, 2M)     Wcat   bf16  [COUT*KTOT]
  //   [4M, 4M+64M)   xT     bf16  [B*T*CIN]
  char* ws = (char*)d_ws;
  int*    idxP = (int*)ws;
  int*    idxF = (int*)(ws + (size_t)B_ * T_ * 4);
  bf16_t* Wcat = (bf16_t*)(ws + 2 * (size_t)B_ * T_ * 4);
  bf16_t* xT   = (bf16_t*)(ws + (size_t)4 * 1024 * 1024);

  build_indices<<<(B_ * T_ + 255) / 256, 256, 0, stream>>>(d, idxP, idxF);

  dim3 gT(T_ / 32, CIN / 32, B_);
  transpose_convert<<<gT, 256, 0, stream>>>(x, xT);

  build_wcat<<<(COUT * KTOT + 255) / 256, 256, 0, stream>>>(W0, W1, W2, Wcat);

  dim3 gG(T_ / 128, COUT / 128, B_);
  adaptive_conv_wmma<<<gG, 256, 0, stream>>>(xT, Wcat, b0, idxP, idxF, out);
}
